// SelfAttHeadLevel1_51969104281923
// MI455X (gfx1250) — matrix operand order repaired and checked by reading, not compiled, not run on hardware
//
#include <hip/hip_runtime.h>
#include <stdint.h>

// Problem sizes (fixed by the reference)
#define BATCH 4
#define SEQ   4096
#define EMBD  512
#define DKD   64

typedef __bf16 v16bf __attribute__((ext_vector_type(16)));
typedef float  v8f   __attribute__((ext_vector_type(8)));

union BfOp { v16bf v; uint4 q[2]; unsigned int u[8]; };

// ---- f32 -> bf16 helpers: native cvt_pk when available, RNE bit-ops fallback
__device__ __forceinline__ unsigned int bf16_bits(float f) {
  unsigned int u = __float_as_uint(f);
  u += 0x7FFFu + ((u >> 16) & 1u);
  return u >> 16;
}
__device__ __forceinline__ unsigned int pack_bf16(float lo, float hi) {
#if __has_builtin(__builtin_amdgcn_cvt_pk_bf16_f32)
  typedef __bf16 v2bf __attribute__((ext_vector_type(2)));
  v2bf r = __builtin_amdgcn_cvt_pk_bf16_f32(lo, hi);
  return __builtin_bit_cast(unsigned int, r);
#else
  unsigned int a = __float_as_uint(lo), b = __float_as_uint(hi);
  a += 0x7FFFu + ((a >> 16) & 1u);
  b += 0x7FFFu + ((b >> 16) & 1u);
  return (a >> 16) | (b & 0xFFFF0000u);
#endif
}
__device__ __forceinline__ float fast_exp2(float x) {
#if __has_builtin(__builtin_amdgcn_exp2f)
  return __builtin_amdgcn_exp2f(x);
#else
  return exp2f(x);
#endif
}

// ---------------------------------------------------------------------------
// Kernel 1: weights -> bf16, transposed (W^T[out][in]) for contiguous operands
// ---------------------------------------------------------------------------
__global__ void weight_prep(const float* __restrict__ wq, const float* __restrict__ wk,
                            const float* __restrict__ wv,
                            unsigned short* __restrict__ wqT,
                            unsigned short* __restrict__ wkT,
                            unsigned short* __restrict__ wvT) {
  int idx = blockIdx.x * blockDim.x + threadIdx.x;
  if (idx < DKD * EMBD) {
    int o = idx / EMBD, d = idx % EMBD;
    wqT[idx] = (unsigned short)bf16_bits(wq[d * DKD + o]);
  } else if (idx < 2 * DKD * EMBD) {
    int i = idx - DKD * EMBD;
    int o = i / EMBD, d = i % EMBD;
    wkT[i] = (unsigned short)bf16_bits(wk[d * DKD + o]);
  } else if (idx < 2 * DKD * EMBD + EMBD * EMBD) {
    int i = idx - 2 * DKD * EMBD;
    int o = i / EMBD, d = i % EMBD;
    wvT[i] = (unsigned short)bf16_bits(wv[d * EMBD + o]);
  }
}

// ---------------------------------------------------------------------------
// Kernel 2: E -> bf16 once (no inline cvt VALU in the GEMMs, no redundant
// per-wave conversion of shared operands).
// ---------------------------------------------------------------------------
__global__ void e_prep(const float* __restrict__ E, unsigned int* __restrict__ Ebf2) {
  int i = blockIdx.x * blockDim.x + threadIdx.x;   // one float4 -> uint2 (4 bf16)
  const int nquads = BATCH * SEQ * EMBD / 4;
  if (i < nquads) {
    float4 f = ((const float4*)E)[i];
    uint2 o;
    o.x = pack_bf16(f.x, f.y);
    o.y = pack_bf16(f.z, f.w);
    ((uint2*)Ebf2)[i] = o;
  }
}

// ---------------------------------------------------------------------------
// Kernel 3: Q = (E*Wq + bq) * (1/8 * log2 e)   [softmax scale folded into Q],
//           K =  E*Wk + bk    (bf16 row-major [B*SEQ][64])
// Block = 16 E-rows; 8 waves: 0-3 -> Q col-tiles, 4-7 -> K col-tiles.
// ---------------------------------------------------------------------------
__global__ __launch_bounds__(256) void qk_proj(
    const unsigned short* __restrict__ Ebf,
    const unsigned short* __restrict__ wqT, const unsigned short* __restrict__ wkT,
    const float* __restrict__ bq, const float* __restrict__ bk,
    unsigned short* __restrict__ Qbf, unsigned short* __restrict__ Kbf) {
  const int rt   = blockIdx.x;          // row-tile over B*SEQ/16
  const int lane = threadIdx.x & 31;
  const int wid  = threadIdx.x >> 5;
  const int kh   = lane >> 4;
  const int lm   = lane & 15;
  const bool isQ = wid < 4;
  const int ct   = isQ ? wid : (wid - 4);
  const unsigned short* WT = isQ ? wqT : wkT;
  const float* bias        = isQ ? bq  : bk;
  unsigned short* Obf      = isQ ? Qbf : Kbf;
  const float postscale    = isQ ? (0.125f * 1.44269504088896340736f) : 1.0f;
  const int colg = ct * 16 + lm;        // output column (C-layout: N = lane)

  v8f acc;
  #pragma unroll
  for (int i = 0; i < 8; ++i) acc[i] = 0.0f;

  for (int ks = 0; ks < EMBD / 32; ++ks) {
    const int ds = ks * 32;
    BfOp a;   // A operand: Ebf[row][{ds+kh*8..+7, ds+16+kh*8..+7}]
    const unsigned short* er = Ebf + (size_t)(rt * 16 + lm) * EMBD + ds + kh * 8;
    a.q[0] = *(const uint4*)(er);
    a.q[1] = *(const uint4*)(er + 16);
    BfOp bm;  // B operand: lane n = column colg, k = {ds+kh*8..+7, ds+16+kh*8..+7}
    const unsigned short* wr = WT + (size_t)colg * EMBD + ds + kh * 8;
    bm.q[0] = *(const uint4*)(wr);
    bm.q[1] = *(const uint4*)(wr + 16);
    acc = __builtin_amdgcn_wmma_f32_16x16x32_bf16(false, a.v, false, bm.v,
                                                  (short)0, acc, false, false);
  }
  const float bv = bias[colg];
  #pragma unroll
  for (int i = 0; i < 8; ++i) {
    const int r = rt * 16 + i + 8 * kh;   // C-layout: M = vgpr + 8*half
    Obf[(size_t)r * DKD + colg] = (unsigned short)bf16_bits((acc[i] + bv) * postscale);
  }
}

// ---------------------------------------------------------------------------
// Kernel 4: V^T = Wv^T * E^T + bv  (bf16 [B][EMBD][SEQ], V stored transposed
// so the attention PV A-operand is a contiguous per-lane read).
// ---------------------------------------------------------------------------
__global__ __launch_bounds__(256) void v_proj(
    const unsigned short* __restrict__ Ebf,
    const unsigned short* __restrict__ wvT, const float* __restrict__ bv,
    unsigned short* __restrict__ VTbf) {
  const int blk  = blockIdx.x;                // over B*SEQ/16
  const int bIdx = blk / (SEQ / 16);
  const int rt   = blk % (SEQ / 16);
  const int lane = threadIdx.x & 31;
  const int wid  = threadIdx.x >> 5;
  const int kh   = lane >> 4;
  const int lm   = lane & 15;

  v8f acc[4];
  #pragma unroll
  for (int t = 0; t < 4; ++t)
    #pragma unroll
    for (int i = 0; i < 8; ++i) acc[t][i] = 0.0f;

  for (int ks = 0; ks < EMBD / 32; ++ks) {
    const int ds = ks * 32;
    BfOp bm;  // B operand (E^T): lane n = column = E row, k = {ds+kh*8..+7, ds+16+kh*8..+7}
    const unsigned short* er = Ebf + (size_t)(blk * 16 + lm) * EMBD + ds + kh * 8;
    bm.q[0] = *(const uint4*)(er);
    bm.q[1] = *(const uint4*)(er + 16);
    #pragma unroll
    for (int t = 0; t < 4; ++t) {
      BfOp a;   // A operand: Wv^T row o = wid*64 + t*16 + lm
      const unsigned short* wr =
          wvT + (size_t)(wid * 64 + t * 16 + lm) * EMBD + ds + kh * 8;
      a.q[0] = *(const uint4*)(wr);
      a.q[1] = *(const uint4*)(wr + 16);
      acc[t] = __builtin_amdgcn_wmma_f32_16x16x32_bf16(false, a.v, false, bm.v,
                                                       (short)0, acc[t], false, false);
    }
  }
  const int rn = rt * 16 + lm;                // row within batch
  #pragma unroll
  for (int t = 0; t < 4; ++t)
    #pragma unroll
    for (int i = 0; i < 8; ++i) {
      const int og = wid * 64 + t * 16 + i + 8 * kh;
      VTbf[((size_t)bIdx * EMBD + og) * SEQ + rn] =
          (unsigned short)bf16_bits(acc[t][i] + bv[og]);
    }
}

// ---------------------------------------------------------------------------
// Kernel 5: causal flash attention, transposed formulation.
// Block = (batch, 16-query tile); 4 waves x 128 output columns each.
// 64 keys per softmax update: 4 S^T tiles (8 WMMA), ONE m/l update + ONE
// accumulator rescale (64 muls amortized over 64 keys), two P^T operands
// built with 16 dword shfl_xor, 16 PV WMMAs. Lane = query in all C-layouts,
// so softmax is lane-local + one shfl. No LDS, no barriers.
// ---------------------------------------------------------------------------
__global__ __launch_bounds__(128) void attn_fwd(
    const unsigned short* __restrict__ Qbf,
    const unsigned short* __restrict__ Kbf,
    const unsigned short* __restrict__ VTbf,
    float* __restrict__ out) {
  const int b    = blockIdx.x >> 8;           // / (SEQ/16)
  const int qt   = blockIdx.x & 255;
  const int lane = threadIdx.x & 31;
  const int wid  = threadIdx.x >> 5;          // 0..3
  const int kh   = lane >> 4;
  const int lm   = lane & 15;
  const int qg   = qt * 16 + lm;              // this lane's query row

  const unsigned short* Qb = Qbf  + (size_t)b * SEQ * DKD;
  const unsigned short* Kb = Kbf  + (size_t)b * SEQ * DKD;
  const unsigned short* Vb = VTbf + (size_t)b * EMBD * SEQ;

  // Q^T B-operands (d 0..31, 32..63); Q already carries 1/8*log2e scale.
  BfOp qop0, qop1;
  {
    const unsigned short* qr = Qb + (size_t)qg * DKD + kh * 16;
    qop0.q[0] = *(const uint4*)(qr);
    qop0.q[1] = *(const uint4*)(qr + 8);
    qop1.q[0] = *(const uint4*)(qr + 32);
    qop1.q[1] = *(const uint4*)(qr + 40);
  }

  v8f acc[8];                                  // O^T tiles: M=col, N=query(lane)
  #pragma unroll
  for (int t = 0; t < 8; ++t)
    #pragma unroll
    for (int i = 0; i < 8; ++i) acc[t][i] = 0.0f;

  float m = -1e30f, l = 0.0f;
  // 64-key chunks; last chunk start is always <= qt*16, so every query row
  // has at least one unmasked key in every processed chunk's tile row range.
  const int nchunk = ((qt * 16 + 15) >> 6) + 1;

  for (int kc = 0; kc < nchunk; ++kc) {
    const int kb = kc * 64;
    if (kc + 1 < nchunk)
      __builtin_prefetch(Kb + (size_t)(kb + 64 + lm) * DKD, 0, 3);

    // ---- S^T tiles: T{tt} = keys kb+16*tt .. kb+16*tt+15 ----
    v8f st[4];
    #pragma unroll
    for (int tt = 0; tt < 4; ++tt)
      #pragma unroll
      for (int i = 0; i < 8; ++i) st[tt][i] = 0.0f;
    #pragma unroll
    for (int s = 0; s < 2; ++s) {
      const v16bf qv = (s == 0) ? qop0.v : qop1.v;
      #pragma unroll
      for (int tt = 0; tt < 4; ++tt) {
        BfOp ko;  // A operand: K row = kb + tt*16 + lm (contiguous bf16)
        const unsigned short* kr =
            Kb + (size_t)(kb + tt * 16 + lm) * DKD + s * 32 + kh * 16;
        ko.q[0] = *(const uint4*)(kr);
        ko.q[1] = *(const uint4*)(kr + 8);
        st[tt] = __builtin_amdgcn_wmma_f32_16x16x32_bf16(false, ko.v, false, qv,
                                                         (short)0, st[tt], false, false);
      }
    }

    // ---- causal mask (scores already in log2 units) ----
    if (kb + 63 > qt * 16) {                  // chunk touches the diagonal
      #pragma unroll
      for (int tt = 0; tt < 4; ++tt)
        #pragma unroll
        for (int i = 0; i < 8; ++i) {
          const int kg = kb + tt * 16 + i + 8 * kh;
          st[tt][i] = (kg <= qg) ? st[tt][i] : -1e30f;
        }
    }

    // ---- one online-softmax update per 64 keys ----
    float mloc = -1e30f;
    #pragma unroll
    for (int tt = 0; tt < 4; ++tt)
      #pragma unroll
      for (int i = 0; i < 8; ++i) mloc = fmaxf(mloc, st[tt][i]);
    const float mcnk  = fmaxf(mloc, __shfl_xor(mloc, 16));
    const float mnew  = fmaxf(m, mcnk);
    const float alpha = fast_exp2(m - mnew);
    float sloc = 0.0f;
    #pragma unroll
    for (int tt = 0; tt < 4; ++tt)
      #pragma unroll
      for (int i = 0; i < 8; ++i) {
        const float p = fast_exp2(st[tt][i] - mnew);
        st[tt][i] = p;
        sloc += p;
      }
    l = l * alpha + sloc + __shfl_xor(sloc, 16);
    m = mnew;

    // ---- two P^T B-operands: pack bf16 pairs first, then dword shfls ----
    BfOp pop01, pop23;
    #pragma unroll
    for (int j = 0; j < 4; ++j) {
      const unsigned int a0 = pack_bf16(st[0][2 * j], st[0][2 * j + 1]);
      const unsigned int a1 = pack_bf16(st[1][2 * j], st[1][2 * j + 1]);
      const unsigned int a2 = pack_bf16(st[2][2 * j], st[2][2 * j + 1]);
      const unsigned int a3 = pack_bf16(st[3][2 * j], st[3][2 * j + 1]);
      const unsigned int h0 = __shfl_xor(a0, 16);
      const unsigned int h1 = __shfl_xor(a1, 16);
      const unsigned int h2 = __shfl_xor(a2, 16);
      const unsigned int h3 = __shfl_xor(a3, 16);
      pop01.u[j]     = kh ? h1 : a0;   // keys  0-7  | 16-23
      pop01.u[4 + j] = kh ? a1 : h0;   // keys  8-15 | 24-31
      pop23.u[j]     = kh ? h3 : a2;   // keys 32-39 | 48-55
      pop23.u[4 + j] = kh ? a3 : h2;   // keys 40-47 | 56-63
    }

    // ---- rescale accumulators once per 64 keys ----
    #pragma unroll
    for (int t = 0; t < 8; ++t)
      #pragma unroll
      for (int i = 0; i < 8; ++i) acc[t][i] *= alpha;

    // ---- O^T += V^T * P^T : 2 k-steps of 32 keys per column tile ----
    #pragma unroll
    for (int t = 0; t < 8; ++t) {
      const unsigned short* vr =
          Vb + (size_t)(wid * 128 + t * 16 + lm) * SEQ + kb + kh * 16;
      BfOp v0, v1;
      v0.q[0] = *(const uint4*)(vr);
      v0.q[1] = *(const uint4*)(vr + 8);
      v1.q[0] = *(const uint4*)(vr + 32);
      v1.q[1] = *(const uint4*)(vr + 40);
      acc[t] = __builtin_amdgcn_wmma_f32_16x16x32_bf16(false, v0.v, false, pop01.v,
                                                       (short)0, acc[t], false, false);
      acc[t] = __builtin_amdgcn_wmma_f32_16x16x32_bf16(false, v1.v, false, pop23.v,
                                                       (short)0, acc[t], false, false);
    }
  }

  // ---- finalize: divide by row sum (per-lane), store O[q][c] fp32 ----
  const float invl = 1.0f / l;
  float* ob = out + ((size_t)b * SEQ + qg) * EMBD + wid * 128;
  #pragma unroll
  for (int t = 0; t < 8; ++t)
    #pragma unroll
    for (int i = 0; i < 8; ++i)
      ob[t * 16 + i + 8 * kh] = acc[t][i] * invl;
}

// ---------------------------------------------------------------------------
// Workspace layout (bytes): ~36.7 MB total
// ---------------------------------------------------------------------------
#define WS_WQT 0u
#define WS_WKT 65536u
#define WS_WVT 131072u
#define WS_EBF 655360u
#define WS_Q   17432576u
#define WS_K   19529728u
#define WS_VT  21626880u
// end: 38404096

extern "C" void kernel_launch(void* const* d_in, const int* in_sizes, int n_in,
                              void* d_out, int out_size, void* d_ws, size_t ws_size,
                              hipStream_t stream) {
  const float* E  = (const float*)d_in[0];
  // d_in[1] = causal mask (tril) -- implemented analytically, not read
  const float* wq = (const float*)d_in[2];
  const float* bq = (const float*)d_in[3];
  const float* wk = (const float*)d_in[4];
  const float* bk = (const float*)d_in[5];
  const float* wv = (const float*)d_in[6];
  const float* bv = (const float*)d_in[7];
  float* out = (float*)d_out;

  char* ws = (char*)d_ws;
  unsigned short* wqT  = (unsigned short*)(ws + WS_WQT);
  unsigned short* wkT  = (unsigned short*)(ws + WS_WKT);
  unsigned short* wvT  = (unsigned short*)(ws + WS_WVT);
  unsigned short* Ebf  = (unsigned short*)(ws + WS_EBF);
  unsigned short* Qbf  = (unsigned short*)(ws + WS_Q);
  unsigned short* Kbf  = (unsigned short*)(ws + WS_K);
  unsigned short* VTbf = (unsigned short*)(ws + WS_VT);

  const int prep_elems = 2 * DKD * EMBD + EMBD * EMBD;   // 327680
  weight_prep<<<(prep_elems + 255) / 256, 256, 0, stream>>>(wq, wk, wv, wqT, wkT, wvT);

  const int equads = BATCH * SEQ * EMBD / 4;             // 2097152
  e_prep<<<(equads + 255) / 256, 256, 0, stream>>>(E, (unsigned int*)Ebf);

  qk_proj<<<(BATCH * SEQ) / 16, 256, 0, stream>>>(Ebf, wqT, wkT, bq, bk, Qbf, Kbf);
  v_proj<<<(BATCH * SEQ) / 16, 256, 0, stream>>>(Ebf, wvT, bv, VTbf);
  attn_fwd<<<BATCH * (SEQ / 16), 128, 0, stream>>>(Qbf, Kbf, VTbf, out);
}
